// AOP_NN_49168785604711
// MI455X (gfx1250) — compile-verified, matrix operand order checked by reading, not run on hardware
//
#include <hip/hip_runtime.h>
#include <hip/hip_bf16.h>
#include <math.h>

// Problem constants (from reference)
#define BB    2048
#define DIN   64
#define NM0   8192
#define NM1   1024
#define HH    8
#define FANIN 8
#define BN_EPS 1e-5f

typedef __attribute__((ext_vector_type(16))) __bf16 v16bf;
typedef __attribute__((ext_vector_type(8)))  float  v8f;

// WMMA 16-bit fragment viewed as 8 dwords (each dword = 2 consecutive-K bf16)
union BF16W { v16bf v; unsigned int u[8]; };

// pack two f32 -> two bf16 (round-to-nearest) in one dword: [b | a]
__device__ __forceinline__ unsigned int pack2bf(float a, float b) {
  unsigned int ua = __builtin_bit_cast(unsigned int, a) + 0x8000u;
  unsigned int ub = __builtin_bit_cast(unsigned int, b) + 0x8000u;
  return (ua >> 16) | (ub & 0xFFFF0000u);
}

// Hardware tanh (CDNA5 v_tanh_f32 transcendental)
__device__ __forceinline__ float fast_tanh(float x) {
#if __has_builtin(__builtin_amdgcn_tanhf)
  return __builtin_amdgcn_tanhf(x);
#elif __has_builtin(__builtin_amdgcn_tanh_f32)
  return __builtin_amdgcn_tanh_f32(x);
#else
  float r;
  // TRANS op: one independent op (v_nop) before result use, per ISA 5.7.4
  asm volatile("v_tanh_f32 %0, %1\n\tv_nop" : "=v"(r) : "v"(x));
  return r;
#endif
}

// Base K (even) of dword j of a CDNA5 16-bit A/B fragment, per ISA 7.12.2:
// VGPR j<4 -> K=2j (+8 hi lane half), VGPR j>=4 -> K=16+2(j-4) (+8 hi half).
__device__ __forceinline__ int kpair(int j, int laneHi) {
  return ((j & 4) ? 16 : 0) + 2 * (j & 3) + (laneHi ? 8 : 0);
}

// ---------------------------------------------------------------------------
// Layer 0: one WG owns 2 MIE terms (16 WMMA output cols) x all 2048 rows.
// GEMM(bf16 WMMA, f32 acc) + tanh -> LDS; batch BN; aux head + batch BN.
// ---------------------------------------------------------------------------
__global__ __launch_bounds__(128, 1)
void mie_layer0(const float* __restrict__ x,   const float* __restrict__ W0,
                const float* __restrict__ b0,  const float* __restrict__ g0,
                const float* __restrict__ be0, const float* __restrict__ Wa0,
                const float* __restrict__ ba0, const float* __restrict__ ga0,
                const float* __restrict__ bea0,
                float* __restrict__ aux0, float* __restrict__ t0g) {
  extern __shared__ float lds[];
  float*          smT   = lds;                         // [2048][16] h0 -> t0
  float*          smA   = lds + 32768;                 // [2048][4]  a0
  unsigned short* smXh  = (unsigned short*)(lds + 40960); // [4][16][64] bf16 x
  float*          red   = lds + 43008;                 // [128][2]
  float*          smSc  = lds + 43264;                 // [16]
  float*          smSh  = lds + 43280;                 // [16]
  float*          smAS  = lds + 43296;                 // [4]
  float*          smASh = lds + 43300;                 // [4]

  const int tid    = threadIdx.x;
  const int wave   = tid >> 5;
  const int lane   = tid & 31;
  const int laneHi = lane >> 4;
  const int col    = lane & 15;          // WMMA N col (B) == M row-in-tile (A)
  const int termBase = blockIdx.x * 2;

  // B fragments: W0 of 2 terms packed as 64x16, N = term_local*8 + h.
  // Element (k,n) = W0[term][k][h]; consecutive K stride = HH floats.
  v16bf bfrag[2];
#pragma unroll
  for (int c = 0; c < 2; ++c) {
    BF16W u;
#pragma unroll
    for (int j = 0; j < 8; ++j) {
      int k = kpair(j, laneHi) + 32 * c;
      const float* wp = W0 + (size_t)(termBase + (col >> 3)) * (DIN * HH)
                           + (size_t)k * HH + (col & 7);
      u.u[j] = pack2bf(wp[0], wp[HH]);
    }
    bfrag[c] = u.v;
  }
  const float bias = b0[(termBase + (col >> 3)) * HH + (col & 7)];

  float sum = 0.f, ssq = 0.f;
  const int rBase0 = wave * (BB / 4);
  for (int it = 0; it < (BB / 4) / 16; ++it) {
    const int rBase = rBase0 + it * 16;
    // stage 16 rows of x (contiguous 1024 floats) -> bf16-packed LDS
    const float4* src = (const float4*)(x + (size_t)rBase * DIN);
    uint2* dst = (uint2*)(smXh + wave * 1024);
#pragma unroll
    for (int i = 0; i < 8; ++i) {
      float4 f = src[lane + 32 * i];
      uint2 p; p.x = pack2bf(f.x, f.y); p.y = pack2bf(f.z, f.w);
      dst[lane + 32 * i] = p;
    }
    if (it + 1 < (BB / 4) / 16)
      __builtin_prefetch(x + (size_t)(rBase + 16) * DIN + lane * 16, 0, 3);

    v8f acc = {};
#pragma unroll
    for (int c = 0; c < 2; ++c) {
      BF16W a;
#pragma unroll
      for (int j = 0; j < 8; ++j)
        a.u[j] = *(const unsigned int*)(smXh + wave * 1024 + col * DIN
                                        + kpair(j, laneHi) + 32 * c);
      acc = __builtin_amdgcn_wmma_f32_16x16x32_bf16(false, a.v, false, bfrag[c],
                                                    (short)0, acc, false, false);
    }
#pragma unroll
    for (int v = 0; v < 8; ++v) {          // C/D: lanes 0-15 -> M=v, 16-31 -> M=v+8
      int row = rBase + (laneHi ? v + 8 : v);
      float hv = fast_tanh(acc[v] + bias);
      sum += hv; ssq += hv * hv;
      smT[row * 16 + col] = hv;
    }
  }

  __syncthreads();
  red[tid * 2] = sum; red[tid * 2 + 1] = ssq;
  __syncthreads();
  if (tid < 16) {                          // per-column batch stats (16 cols)
    float s = 0.f, q = 0.f;
#pragma unroll
    for (int w = 0; w < 4; ++w) {
      s += red[(w * 32 + tid) * 2]     + red[(w * 32 + 16 + tid) * 2];
      q += red[(w * 32 + tid) * 2 + 1] + red[(w * 32 + 16 + tid) * 2 + 1];
    }
    float mean = s * (1.f / BB);
    float var  = fmaxf(q * (1.f / BB) - mean * mean, 0.f);
    float gam  = g0[(termBase + (tid >> 3)) * HH + (tid & 7)];
    float bet  = be0[(termBase + (tid >> 3)) * HH + (tid & 7)];
    float sc   = gam * rsqrtf(var + BN_EPS);
    smSc[tid] = sc; smSh[tid] = bet - mean * sc;
  }
  __syncthreads();
  for (int idx = tid; idx < BB * 16; idx += 128) {   // normalize + stream t0
    int c = idx & 15, r = idx >> 4;
    float t = smT[idx] * smSc[c] + smSh[c];
    smT[idx] = t;
    t0g[(size_t)r * (NM0 * HH) + blockIdx.x * 16 + c] = t;
  }
  __syncthreads();
  {                                        // aux head: K=8, N=2 per term (VALU)
    const int ac = tid & 3, rg = tid >> 2;
    const int tl = ac >> 1, kk = ac & 1;
    float wa[8];
#pragma unroll
    for (int h = 0; h < 8; ++h)
      wa[h] = Wa0[(size_t)(termBase + tl) * (HH * 2) + h * 2 + kk];
    const float ab = ba0[(termBase + tl) * 2 + kk];
    float s = 0.f, q = 0.f;
    for (int r = rg * 64; r < rg * 64 + 64; ++r) {
      float a = ab;
#pragma unroll
      for (int h = 0; h < 8; ++h) a += smT[r * 16 + tl * 8 + h] * wa[h];
      a = fast_tanh(a);
      smA[r * 4 + ac] = a;
      s += a; q += a * a;
    }
    red[tid * 2] = s; red[tid * 2 + 1] = q;
  }
  __syncthreads();
  if (tid < 4) {
    float s = 0.f, q = 0.f;
    for (int rg = 0; rg < 32; ++rg) { s += red[(rg * 4 + tid) * 2]; q += red[(rg * 4 + tid) * 2 + 1]; }
    float mean = s * (1.f / BB);
    float var  = fmaxf(q * (1.f / BB) - mean * mean, 0.f);
    float gam  = ga0[(termBase + (tid >> 1)) * 2 + (tid & 1)];
    float bet  = bea0[(termBase + (tid >> 1)) * 2 + (tid & 1)];
    float sc   = gam * rsqrtf(var + BN_EPS);
    smAS[tid] = sc; smASh[tid] = bet - mean * sc;
  }
  __syncthreads();
  for (int idx = tid; idx < BB * 4; idx += 128) {
    int c = idx & 3, r = idx >> 2;
    aux0[(size_t)r * (NM0 * 2) + blockIdx.x * 4 + c] = smA[idx] * smAS[c] + smASh[c];
  }
}

// ---------------------------------------------------------------------------
// Layer 1: one WG owns 2 root terms. Block-diagonal B: K=128 (2 roots x 64
// children), N=16 (2 roots x 8 hidden) so all WMMA output columns are live.
// ---------------------------------------------------------------------------
__global__ __launch_bounds__(128, 1)
void root_layer1(const float* __restrict__ t0g, const float* __restrict__ W1,
                 const float* __restrict__ b1,  const float* __restrict__ g1,
                 const float* __restrict__ be1, const float* __restrict__ Wa1,
                 const float* __restrict__ ba1, const float* __restrict__ ga1,
                 const float* __restrict__ bea1,
                 float* __restrict__ aux1, float* __restrict__ t1g) {
  extern __shared__ float lds[];
  float*          smT   = lds;                         // [2048][16] h1 -> t1
  float*          smA   = lds + 32768;                 // [2048][4]
  unsigned short* smXh  = (unsigned short*)(lds + 40960); // [4][16][128] bf16 t0
  float*          red   = lds + 45056;                 // [128][2]
  float*          smSc  = lds + 45312;
  float*          smSh  = lds + 45328;
  float*          smAS  = lds + 45344;
  float*          smASh = lds + 45348;

  const int tid    = threadIdx.x;
  const int wave   = tid >> 5;
  const int lane   = tid & 31;
  const int laneHi = lane >> 4;
  const int col    = lane & 15;
  const int rootBase = blockIdx.x * 2;

  // block-diagonal B: K rows [0,64) -> root0 cols 0-7, [64,128) -> root1 cols 8-15
  v16bf bfrag[4];
  const int rn = col >> 3, hsel = col & 7;
#pragma unroll
  for (int c = 0; c < 4; ++c) {
    BF16W u;
#pragma unroll
    for (int j = 0; j < 8; ++j) {
      int k = kpair(j, laneHi) + 32 * c;       // even; k,k+1 share the 64-block
      if ((k >> 6) == rn) {
        const float* wp = W1 + (size_t)(rootBase + rn) * (FANIN * HH * HH)
                             + (size_t)(k & 63) * HH + hsel;
        u.u[j] = pack2bf(wp[0], wp[HH]);
      } else {
        u.u[j] = 0u;
      }
    }
    bfrag[c] = u.v;
  }
  const float bias = b1[(rootBase + rn) * HH + hsel];

  float sum = 0.f, ssq = 0.f;
  const int rBase0 = wave * (BB / 4);
  for (int it = 0; it < (BB / 4) / 16; ++it) {
    const int rBase = rBase0 + it * 16;
#pragma unroll
    for (int r = 0; r < 16; ++r) {           // 128 contiguous floats per row
      const float4* src = (const float4*)(t0g + (size_t)(rBase + r) * (NM0 * HH)
                                              + (size_t)blockIdx.x * 128);
      float4 f = src[lane];
      uint2 p; p.x = pack2bf(f.x, f.y); p.y = pack2bf(f.z, f.w);
      ((uint2*)(smXh + wave * 2048 + r * 128))[lane] = p;
    }
    if (it + 1 < (BB / 4) / 16)
      __builtin_prefetch(t0g + (size_t)(rBase + 16) * (NM0 * HH)
                             + (size_t)blockIdx.x * 128 + lane * 4, 0, 3);

    v8f acc = {};
#pragma unroll
    for (int c = 0; c < 4; ++c) {
      BF16W a;
#pragma unroll
      for (int j = 0; j < 8; ++j)
        a.u[j] = *(const unsigned int*)(smXh + wave * 2048 + col * 128
                                        + kpair(j, laneHi) + 32 * c);
      acc = __builtin_amdgcn_wmma_f32_16x16x32_bf16(false, a.v, false, bfrag[c],
                                                    (short)0, acc, false, false);
    }
#pragma unroll
    for (int v = 0; v < 8; ++v) {
      int row = rBase + (laneHi ? v + 8 : v);
      float hv = fast_tanh(acc[v] + bias);
      sum += hv; ssq += hv * hv;
      smT[row * 16 + col] = hv;
    }
  }

  __syncthreads();
  red[tid * 2] = sum; red[tid * 2 + 1] = ssq;
  __syncthreads();
  if (tid < 16) {
    float s = 0.f, q = 0.f;
#pragma unroll
    for (int w = 0; w < 4; ++w) {
      s += red[(w * 32 + tid) * 2]     + red[(w * 32 + 16 + tid) * 2];
      q += red[(w * 32 + tid) * 2 + 1] + red[(w * 32 + 16 + tid) * 2 + 1];
    }
    float mean = s * (1.f / BB);
    float var  = fmaxf(q * (1.f / BB) - mean * mean, 0.f);
    float gam  = g1[(rootBase + (tid >> 3)) * HH + (tid & 7)];
    float bet  = be1[(rootBase + (tid >> 3)) * HH + (tid & 7)];
    float sc   = gam * rsqrtf(var + BN_EPS);
    smSc[tid] = sc; smSh[tid] = bet - mean * sc;
  }
  __syncthreads();
  for (int idx = tid; idx < BB * 16; idx += 128) {
    int c = idx & 15, r = idx >> 4;
    float t = smT[idx] * smSc[c] + smSh[c];
    smT[idx] = t;
    t1g[(size_t)r * (NM1 * HH) + blockIdx.x * 16 + c] = t;
  }
  __syncthreads();
  {
    const int ac = tid & 3, rg = tid >> 2;
    const int tl = ac >> 1, kk = ac & 1;
    float wa[8];
#pragma unroll
    for (int h = 0; h < 8; ++h)
      wa[h] = Wa1[(size_t)(rootBase + tl) * (HH * 2) + h * 2 + kk];
    const float ab = ba1[(rootBase + tl) * 2 + kk];
    float s = 0.f, q = 0.f;
    for (int r = rg * 64; r < rg * 64 + 64; ++r) {
      float a = ab;
#pragma unroll
      for (int h = 0; h < 8; ++h) a += smT[r * 16 + tl * 8 + h] * wa[h];
      a = fast_tanh(a);
      smA[r * 4 + ac] = a;
      s += a; q += a * a;
    }
    red[tid * 2] = s; red[tid * 2 + 1] = q;
  }
  __syncthreads();
  if (tid < 4) {
    float s = 0.f, q = 0.f;
    for (int rg = 0; rg < 32; ++rg) { s += red[(rg * 4 + tid) * 2]; q += red[(rg * 4 + tid) * 2 + 1]; }
    float mean = s * (1.f / BB);
    float var  = fmaxf(q * (1.f / BB) - mean * mean, 0.f);
    float gam  = ga1[(rootBase + (tid >> 1)) * 2 + (tid & 1)];
    float bet  = bea1[(rootBase + (tid >> 1)) * 2 + (tid & 1)];
    float sc   = gam * rsqrtf(var + BN_EPS);
    smAS[tid] = sc; smASh[tid] = bet - mean * sc;
  }
  __syncthreads();
  for (int idx = tid; idx < BB * 4; idx += 128) {
    int c = idx & 3, r = idx >> 2;
    aux1[(size_t)r * (NM1 * 2) + blockIdx.x * 4 + c] = smA[idx] * smAS[c] + smASh[c];
  }
}

// ---------------------------------------------------------------------------
extern "C" void kernel_launch(void* const* d_in, const int* in_sizes, int n_in,
                              void* d_out, int out_size, void* d_ws, size_t ws_size,
                              hipStream_t stream) {
  (void)in_sizes; (void)n_in; (void)d_ws; (void)ws_size; (void)out_size;

  const float* x    = (const float*)d_in[0];
  const float* W0   = (const float*)d_in[1];
  const float* b0   = (const float*)d_in[2];
  const float* g0   = (const float*)d_in[3];
  const float* be0  = (const float*)d_in[4];
  const float* Wa0  = (const float*)d_in[5];
  const float* ba0  = (const float*)d_in[6];
  const float* ga0  = (const float*)d_in[7];
  const float* bea0 = (const float*)d_in[8];
  const float* W1   = (const float*)d_in[9];
  const float* b1   = (const float*)d_in[10];
  const float* g1   = (const float*)d_in[11];
  const float* be1  = (const float*)d_in[12];
  const float* Wa1  = (const float*)d_in[13];
  const float* ba1  = (const float*)d_in[14];
  const float* ga1  = (const float*)d_in[15];
  const float* bea1 = (const float*)d_in[16];

  float* out  = (float*)d_out;                    // tuple (aux0, aux1, t0, t1)
  float* aux0 = out;                              // [2048][8192][2]
  float* aux1 = aux0 + (size_t)BB * NM0 * 2;      // [2048][1024][2]
  float* t0   = aux1 + (size_t)BB * NM1 * 2;      // [2048][8192][8]
  float* t1   = t0   + (size_t)BB * NM0 * HH;     // [2048][1024][8]

  const int lds0 = 43304 * (int)sizeof(float);    // ~169 KB of the 320 KB WGP LDS
  const int lds1 = 45352 * (int)sizeof(float);    // ~177 KB
  hipFuncSetAttribute(reinterpret_cast<const void*>(mie_layer0),
                      hipFuncAttributeMaxDynamicSharedMemorySize, lds0);
  hipFuncSetAttribute(reinterpret_cast<const void*>(root_layer1),
                      hipFuncAttributeMaxDynamicSharedMemorySize, lds1);

  mie_layer0 <<<NM0 / 2, 128, lds0, stream>>>(x, W0, b0, g0, be0, Wa0, ba0, ga0, bea0,
                                              aux0, t0);
  root_layer1<<<NM1 / 2, 128, lds1, stream>>>(t0, W1, b1, g1, be1, Wa1, ba1, ga1, bea1,
                                              aux1, t1);
}
